// StringGNNBilinearModel_6923487281887
// MI455X (gfx1250) — compile-verified, matrix operand order checked by reading, not run on hardware
//
#include <hip/hip_runtime.h>
#include <hip/hip_bf16.h>

// ---------------- problem constants (from reference) ----------------
#define NN     18870      // N_NODES
#define DGN    256        // GNN_DIM
#define NE     1000000    // N_EDGES
#define NB     4096       // BATCH
#define NH     512        // HIDDEN
#define NEX    2048       // HIDDEN*EXPAND
#define NRANK  512        // RANK
#define NCLS   3          // N_CLASSES
#define NG     6640       // N_GENES
#define NBLK   6          // N_BLOCKS

typedef unsigned short u16;
typedef __attribute__((ext_vector_type(16))) __bf16 bf16x16;
typedef __attribute__((ext_vector_type(8)))  float  f32x8;

union FragU { bf16x16 v; unsigned u[8]; };

// fp32 -> bf16 bits, round-to-nearest-even (only used OFF the GEMM hot path)
__device__ __forceinline__ u16 f2bf(float f) {
  unsigned x = __float_as_uint(f);
  return (u16)((x + 0x7FFFu + ((x >> 16) & 1u)) >> 16);
}

__device__ __forceinline__ float gelu_tanh(float x) {
  float x3 = x * x * x;
  return 0.5f * x * (1.0f + tanhf(0.7978845608028654f * (x + 0.044715f * x3)));
}

// CDNA5 async global->LDS copy (ASYNCcnt-tracked), 16 bytes per lane.
// vdst = LDS byte address (u32 VGPR), vaddr = 64-bit global address.
__device__ __forceinline__ void async_cp16(unsigned lds, const void* g) {
  asm volatile("global_load_async_to_lds_b128 %0, %1, off"
               :: "v"(lds), "v"(g) : "memory");
}
__device__ __forceinline__ void wait_async0() {
  asm volatile("s_wait_asynccnt 0x0" ::: "memory");
}

// Load a 16x32 bf16 fragment (A-layout per ISA 7.12.2) from an LDS row.
// lanes 0-15: u[0..3] = K{0,1..6,7}, u[4..7] = K{16..23}
// lanes 16-31: u[0..3] = K{8..15},   u[4..7] = K{24..31}
__device__ __forceinline__ bf16x16 ld_frag(const u16* rp, int hf) {
  FragU f;
#pragma unroll
  for (int i = 0; i < 4; ++i) {
    f.u[i]     = *(const unsigned*)(rp + 2 * i + 8 * hf);
    f.u[4 + i] = *(const unsigned*)(rp + 16 + 2 * i + 8 * hf);
  }
  return f.v;
}

// ---------------- WMMA bf16 GEMM, async-staged, double-buffered ----------------
// C[M,N] = A[M,K] @ B^T + bias (+gelu) (+residual);  A[M,K], B[N,K] bf16.
// Tile 128(M) x 64(N) x 64(K); 256 threads = 8 waves; each wave owns 32x32 of C
// -> 8 v_wmma_f32_16x16x32_bf16 per K-stage. Stages copied with
// global_load_async_to_lds_b128 into the back buffer while computing the front.
#define TM 128
#define TN 64
#define TK 64
#define LROW (TK + 8)   // 72 bf16 = 144B rows: 16B aligned, conflict-free frags

__global__ __launch_bounds__(256) void k_gemm_bf16(
    const u16* __restrict__ A, const u16* __restrict__ B,
    const float* __restrict__ bias, const float* __restrict__ Res,
    float* __restrict__ C, u16* __restrict__ Cb,
    int M, int N, int K, int act) {
  __shared__ u16 As[2][TM * LROW];
  __shared__ u16 Bs[2][TN * LROW];

  const int tid  = threadIdx.x;
  const int wave = tid >> 5;
  const int lane = tid & 31;
  const int hf   = lane >> 4;
  const int l16  = lane & 15;
  const int wm   = (wave >> 1) * 32;   // 0/32/64/96
  const int wn   = (wave & 1) * 32;    // 0/32
  const int m0   = blockIdx.y * TM;
  const int n0   = blockIdx.x * TN;

  f32x8 acc[2][2];
#pragma unroll
  for (int a = 0; a < 2; ++a)
#pragma unroll
    for (int b = 0; b < 2; ++b)
#pragma unroll
      for (int j = 0; j < 8; ++j) acc[a][b][j] = 0.0f;

  // stage one 128x64 A tile + 64x64 B tile (bf16) into LDS buffers via async DMA
  auto stage = [&](int k0, u16* Ad, u16* Bd) {
#pragma unroll
    for (int it = 0; it < 4; ++it) {     // A: 128 rows x 8 chunks of 16B
      int i = it * 256 + tid;
      int r = i >> 3, c = i & 7;
      int gr = m0 + r; gr = (gr < M) ? gr : (M - 1);   // clamp: garbage masked at store
      async_cp16((unsigned)(size_t)(Ad + r * LROW + c * 8),
                 A + (size_t)gr * K + k0 + c * 8);
    }
#pragma unroll
    for (int it = 0; it < 2; ++it) {     // B: 64 rows x 8 chunks of 16B
      int i = it * 256 + tid;
      int r = i >> 3, c = i & 7;
      int gn = n0 + r; gn = (gn < N) ? gn : (N - 1);
      async_cp16((unsigned)(size_t)(Bd + r * LROW + c * 8),
                 B + (size_t)gn * K + k0 + c * 8);
    }
  };

  stage(0, As[0], Bs[0]);
  wait_async0();
  __syncthreads();

  int buf = 0;
  for (int k0 = 0; k0 < K; k0 += TK, buf ^= 1) {
    if (k0 + TK < K) stage(k0 + TK, As[buf ^ 1], Bs[buf ^ 1]);  // prefetch next

    const u16* Ab = As[buf];
    const u16* Bb = Bs[buf];
#pragma unroll
    for (int kf = 0; kf < 2; ++kf) {
      bf16x16 a0 = ld_frag(Ab + (wm + l16) * LROW + kf * 32, hf);
      bf16x16 a1 = ld_frag(Ab + (wm + 16 + l16) * LROW + kf * 32, hf);
      bf16x16 b0 = ld_frag(Bb + (wn + l16) * LROW + kf * 32, hf);
      bf16x16 b1 = ld_frag(Bb + (wn + 16 + l16) * LROW + kf * 32, hf);
      acc[0][0] = __builtin_amdgcn_wmma_f32_16x16x32_bf16(false, a0, false, b0, (short)0, acc[0][0], false, false);
      acc[0][1] = __builtin_amdgcn_wmma_f32_16x16x32_bf16(false, a0, false, b1, (short)0, acc[0][1], false, false);
      acc[1][0] = __builtin_amdgcn_wmma_f32_16x16x32_bf16(false, a1, false, b0, (short)0, acc[1][0], false, false);
      acc[1][1] = __builtin_amdgcn_wmma_f32_16x16x32_bf16(false, a1, false, b1, (short)0, acc[1][1], false, false);
    }
    wait_async0();      // back-buffer copies complete
    __syncthreads();    // everyone done reading front buffer
  }

  // ---- epilogue: bias -> gelu -> residual -> store (fp32 and/or bf16) ----
#pragma unroll
  for (int mf = 0; mf < 2; ++mf)
#pragma unroll
    for (int nf = 0; nf < 2; ++nf)
#pragma unroll
      for (int j = 0; j < 8; ++j) {
        int row = m0 + wm + mf * 16 + hf * 8 + j;
        int col = n0 + wn + nf * 16 + l16;
        if (row < M && col < N) {
          float v = acc[mf][nf][j];
          if (bias) v += bias[col];
          if (act)  v = gelu_tanh(v);
          if (Res)  v += Res[(size_t)row * N + col];
          if (C)  C[(size_t)row * N + col] = v;
          if (Cb) Cb[(size_t)row * N + col] = f2bf(v);
        }
      }
}

// ---------------- conversion kernels (off critical path) ----------------
__global__ void k_cvt(const float* __restrict__ x, u16* __restrict__ y, size_t n) {
  size_t i = (size_t)blockIdx.x * blockDim.x + threadIdx.x;
  if (i < n) y[i] = f2bf(x[i]);
}
// x[K,N] fp32 -> y[N,K] bf16 (pre-transposed B operand)
__global__ void k_cvt_t(const float* __restrict__ x, u16* __restrict__ y, int K, int N) {
  size_t i = (size_t)blockIdx.x * blockDim.x + threadIdx.x;
  if (i < (size_t)N * K) {
    int n = (int)(i / K), k = (int)(i % K);
    y[i] = f2bf(x[(size_t)k * N + n]);
  }
}

// ---------------- GCN helper kernels ----------------
__global__ void k_zero(float* p, size_t n) {
  size_t i = (size_t)blockIdx.x * blockDim.x + threadIdx.x;
  if (i < n) p[i] = 0.0f;
}

__global__ void k_deg(const int* __restrict__ dst, const float* __restrict__ ew,
                      float* __restrict__ deg, int e_cnt) {
  int e = blockIdx.x * blockDim.x + threadIdx.x;
  if (e < e_cnt) atomicAdd(&deg[dst[e]], ew[e]);
}

__global__ void k_invsqrt(const float* __restrict__ deg, float* __restrict__ invd, int n) {
  int i = blockIdx.x * blockDim.x + threadIdx.x;
  if (i < n) invd[i] = rsqrtf(fmaxf(deg[i], 1e-12f));
}

__global__ void k_norm(const int* __restrict__ src, const int* __restrict__ dst,
                       const float* __restrict__ ew, const float* __restrict__ invd,
                       float* __restrict__ nrm, int e_cnt) {
  int e = blockIdx.x * blockDim.x + threadIdx.x;
  if (e < e_cnt) nrm[e] = ew[e] * invd[src[e]] * invd[dst[e]];
}

// acc[dst] += norm * h[src]; 64 lanes per edge, float4 per lane (256 feats)
__global__ __launch_bounds__(256) void k_scatter(
    const float* __restrict__ h, const int* __restrict__ src,
    const int* __restrict__ dst, const float* __restrict__ nrm,
    float* __restrict__ acc, int e_cnt) {
  int e = blockIdx.x * 4 + (threadIdx.x >> 6);
  int t = threadIdx.x & 63;
  if (e >= e_cnt) return;
  int s = src[e], d = dst[e];
  float w = nrm[e];
  float4 v = *(const float4*)(h + (size_t)s * DGN + t * 4);
  float* o = acc + (size_t)d * DGN + t * 4;
  atomicAdd(o + 0, w * v.x);
  atomicAdd(o + 1, w * v.y);
  atomicAdd(o + 2, w * v.z);
  atomicAdd(o + 3, w * v.w);
}

// v = acc + bias[col] + x ; optionally write fp32 y and/or bf16 yb
__global__ void k_bias_res(const float* __restrict__ acc, const float* __restrict__ bias,
                           const float* __restrict__ x, float* __restrict__ y,
                           u16* __restrict__ yb, size_t n) {
  size_t i = (size_t)blockIdx.x * blockDim.x + threadIdx.x;
  if (i < n) {
    float v = acc[i] + bias[i & (DGN - 1)] + x[i];
    if (y)  y[i] = v;
    if (yb) yb[i] = f2bf(v);
  }
}

__global__ void k_gather_rows(const float* __restrict__ x, const int* __restrict__ idx,
                              float* __restrict__ y, int rows) {
  int b = blockIdx.x;
  int t = threadIdx.x;
  if (b < rows) y[(size_t)b * DGN + t] = x[(size_t)idx[b] * DGN + t];
}

// ---------------- layernorm: one wave32 per row, bf16 out ----------------
__global__ void k_layernorm_bf(const float* __restrict__ x, const float* __restrict__ g,
                               const float* __restrict__ b, u16* __restrict__ y,
                               int rows, int dim) {
  int wid  = blockIdx.x * (blockDim.x >> 5) + (threadIdx.x >> 5);
  int lane = threadIdx.x & 31;
  if (wid >= rows) return;
  const float* xr = x + (size_t)wid * dim;
  float s = 0.f;
  for (int j = lane; j < dim; j += 32) s += xr[j];
#pragma unroll
  for (int o = 16; o > 0; o >>= 1) s += __shfl_xor(s, o, 32);
  float mu = s / (float)dim;
  float v = 0.f;
  for (int j = lane; j < dim; j += 32) { float d = xr[j] - mu; v += d * d; }
#pragma unroll
  for (int o = 16; o > 0; o >>= 1) v += __shfl_xor(v, o, 32);
  float inv = rsqrtf(v / (float)dim + 1e-5f);
  u16* yr = y + (size_t)wid * dim;
  for (int j = lane; j < dim; j += 32) yr[j] = f2bf((xr[j] - mu) * inv * g[j] + b[j]);
}

// ---------------- host-side orchestration ----------------
static inline void gemm(const u16* A, const u16* B, const float* bias,
                        const float* Res, float* C, u16* Cb,
                        int M, int N, int K, int act, hipStream_t s) {
  dim3 grid((N + TN - 1) / TN, (M + TM - 1) / TM);
  k_gemm_bf16<<<grid, 256, 0, s>>>(A, B, bias, Res, C, Cb, M, N, K, act);
}

extern "C" void kernel_launch(void* const* d_in, const int* in_sizes, int n_in,
                              void* d_out, int out_size, void* d_ws, size_t ws_size,
                              hipStream_t stream) {
  (void)in_sizes; (void)n_in; (void)out_size; (void)ws_size;

  const float* mid    = (const float*)d_in[0];          // [NN, 256]
  const int*   ei     = (const int*)d_in[1];            // [2, NE]
  const float* ew     = (const float*)d_in[2];          // [NE]
  const int*   pidx   = (const int*)d_in[3];            // [NB]
  const float* gcn6_w = (const float*)d_in[4];
  const float* gcn6_b = (const float*)d_in[5];
  const float* gcn7_w = (const float*)d_in[6];
  const float* gcn7_b = (const float*)d_in[7];
  const float* post_w = (const float*)d_in[8];
  const float* post_b = (const float*)d_in[9];
  const float* ln_in_g = (const float*)d_in[10];
  const float* ln_in_b = (const float*)d_in[11];
  const float* proj_w  = (const float*)d_in[12];
  const float* proj_b  = (const float*)d_in[13];
  const float* blk[NBLK][6];     // {ln_g, ln_b, w1, b1, w2, b2} from index 14
  for (int k = 0; k < NBLK; ++k)
    for (int j = 0; j < 6; ++j) blk[k][j] = (const float*)d_in[14 + 6 * k + j];
  const float* final_g = (const float*)d_in[50];
  const float* final_b = (const float*)d_in[51];
  const float* rank_w  = (const float*)d_in[52];
  const float* rank_b  = (const float*)d_in[53];
  const float* gene    = (const float*)d_in[54];        // [NG, 512] == [N,K]

  const int* src = ei;
  const int* dst = ei + NE;

  // 256B-granular bump allocator over d_ws
  char* base = (char*)d_ws;
  size_t off = 0;
  auto allocB = [&](size_t bytes) -> char* {
    char* p = base + off;
    off += (bytes + 255) & ~(size_t)255;
    return p;
  };
  auto allocF = [&](size_t n) -> float* { return (float*)allocB(n * 4); };
  auto allocH = [&](size_t n) -> u16*   { return (u16*)allocB(n * 2); };

  const size_t nfeat = (size_t)NN * DGN;

  float* deg  = allocF(NN);
  float* invd = allocF(NN);
  float* nrm  = allocF(NE);
  float* hbuf = allocF(nfeat);            // GEMM out / xp; later hosts t2b
  float* acc  = allocF(nfeat);            // scatter accumulator; later hosts rbufb
  float* x1   = allocF(nfeat);
  float* pert = allocF((size_t)NB * DGN);
  float* hb   = allocF((size_t)NB * NH);
  u16* mid_b = allocH(nfeat);
  u16* x1b   = allocH(nfeat);
  u16* x2b   = allocH(nfeat);
  u16* t0b   = allocH((size_t)NB * DGN);
  u16* t1b   = allocH((size_t)NB * NH);
  u16* w6t   = allocH((size_t)DGN * DGN);
  u16* w7t   = allocH((size_t)DGN * DGN);
  u16* postt = allocH((size_t)DGN * DGN);
  u16* projt = allocH((size_t)NH * DGN);
  u16* w1t[NBLK];
  u16* w2t[NBLK];
  for (int k = 0; k < NBLK; ++k) {
    w1t[k] = allocH((size_t)NEX * NH);
    w2t[k] = allocH((size_t)NH * NEX);
  }
  u16* rankt = allocH((size_t)NRANK * NCLS * NH);
  u16* geneb = allocH((size_t)NG * NRANK);
  // overlays (regions free by the time these are used)
  u16* t2b   = (u16*)hbuf;   // [NB,2048] bf16 = 16.8MB <= 19.3MB
  u16* rbufb = (u16*)acc;    // [NB,1536] bf16 = 12.6MB <= 19.3MB

  // ---- weight/operand conversions (independent, issued first) ----
  auto cvt = [&](const float* x, u16* y, size_t n) {
    k_cvt<<<(int)((n + 255) / 256), 256, 0, stream>>>(x, y, n);
  };
  auto cvtT = [&](const float* x, u16* y, int K, int N) {
    size_t n = (size_t)K * N;
    k_cvt_t<<<(int)((n + 255) / 256), 256, 0, stream>>>(x, y, K, N);
  };
  cvt(mid, mid_b, nfeat);
  cvtT(gcn6_w, w6t, DGN, DGN);
  cvtT(gcn7_w, w7t, DGN, DGN);
  cvtT(post_w, postt, DGN, DGN);
  cvtT(proj_w, projt, DGN, NH);
  for (int k = 0; k < NBLK; ++k) {
    cvtT(blk[k][2], w1t[k], NH, NEX);
    cvtT(blk[k][4], w2t[k], NEX, NH);
  }
  cvtT(rank_w, rankt, NH, NRANK * NCLS);
  cvt(gene, geneb, (size_t)NG * NRANK);

  // ---- degree normalization ----
  k_zero<<<(NN + 255) / 256, 256, 0, stream>>>(deg, NN);
  k_deg<<<(NE + 255) / 256, 256, 0, stream>>>(dst, ew, deg, NE);
  k_invsqrt<<<(NN + 255) / 256, 256, 0, stream>>>(deg, invd, NN);
  k_norm<<<(NE + 255) / 256, 256, 0, stream>>>(src, dst, ew, invd, nrm, NE);

  // ---- GCN layer 6 ----
  gemm(mid_b, w6t, nullptr, nullptr, hbuf, nullptr, NN, DGN, DGN, 0, stream);
  k_zero<<<(int)((nfeat + 255) / 256), 256, 0, stream>>>(acc, nfeat);
  k_scatter<<<(NE + 3) / 4, 256, 0, stream>>>(hbuf, src, dst, nrm, acc, NE);
  k_bias_res<<<(int)((nfeat + 255) / 256), 256, 0, stream>>>(acc, gcn6_b, mid, x1, x1b, nfeat);

  // ---- GCN layer 7 ----
  gemm(x1b, w7t, nullptr, nullptr, hbuf, nullptr, NN, DGN, DGN, 0, stream);
  k_zero<<<(int)((nfeat + 255) / 256), 256, 0, stream>>>(acc, nfeat);
  k_scatter<<<(NE + 3) / 4, 256, 0, stream>>>(hbuf, src, dst, nrm, acc, NE);
  k_bias_res<<<(int)((nfeat + 255) / 256), 256, 0, stream>>>(acc, gcn7_b, x1, nullptr, x2b, nfeat);

  // ---- post_mp (xp into hbuf) + pert gather ----
  gemm(x2b, postt, post_b, nullptr, hbuf, nullptr, NN, DGN, DGN, 0, stream);
  k_gather_rows<<<NB, DGN, 0, stream>>>(hbuf, pidx, pert, NB);

  // ---- head: LN -> proj ----
  k_layernorm_bf<<<(NB + 7) / 8, 256, 0, stream>>>(pert, ln_in_g, ln_in_b, t0b, NB, DGN);
  gemm(t0b, projt, proj_b, nullptr, hb, nullptr, NB, NH, DGN, 0, stream);

  // ---- 6 residual MLP blocks ----
  for (int k = 0; k < NBLK; ++k) {
    k_layernorm_bf<<<(NB + 7) / 8, 256, 0, stream>>>(hb, blk[k][0], blk[k][1], t1b, NB, NH);
    gemm(t1b, w1t[k], blk[k][3], nullptr, nullptr, t2b, NB, NEX, NH, /*gelu*/1, stream);
    gemm(t2b, w2t[k], blk[k][5], /*res*/hb, hb, nullptr, NB, NH, NEX, 0, stream);
  }

  // ---- final LN -> rank projection (bf16 out) ----
  k_layernorm_bf<<<(NB + 7) / 8, 256, 0, stream>>>(hb, final_g, final_b, t1b, NB, NH);
  gemm(t1b, rankt, rank_b, nullptr, nullptr, rbufb, NB, NRANK * NCLS, NH, 0, stream);

  // ---- logits: [NB*3,512] @ gene_emb^T -> [NB*3,6640] fp32 (d_out) ----
  gemm(rbufb, geneb, nullptr, nullptr, (float*)d_out, nullptr, NB * NCLS, NG, NRANK, 0, stream);
}
